// AnatomicalBiasedAttention_8976481648778
// MI455X (gfx1250) — matrix operand order repaired and checked by reading, not compiled, not run on hardware
//
#include <hip/hip_runtime.h>
#include <hip/hip_bf16.h>

#define BB      2
#define NQS     2048
#define NKS     2048
#define DMODEL  1024
#define NHEADS  16
#define DHEAD   64

typedef __attribute__((ext_vector_type(16))) __bf16 v16bf;
typedef __attribute__((ext_vector_type(8)))  float  v8f;
typedef __attribute__((ext_vector_type(4)))  float  f32x4;
typedef __attribute__((ext_vector_type(8)))  __bf16 bf16x8;

// ---------------------------------------------------------------------------
// Fragment packing helpers (CDNA5 WMMA 16x16x32 bf16 layouts, wave32)
// A (16x32, MxK): lane 0-15 -> M=lane, K={0..7,16..23}; lane 16-31 -> M=lane-16,
//                 K={8..15,24..31}. Caller passes p = rowbase + k0 + 8*lanehi;
//                 elements are p[0..7] and p[16..23].
// B (32x16, KxN): lane -> N=lane&15; K same as A (p = colbase + k0 + 8*lanehi).
// C (16x16 f32): lane -> N=lane&15; vgpr r -> M = r + 8*lanehi.
// ---------------------------------------------------------------------------
__device__ __forceinline__ v16bf pack_a(const float* p) {
  f32x4 a0 = *(const f32x4*)(p);
  f32x4 a1 = *(const f32x4*)(p + 4);
  f32x4 a2 = *(const f32x4*)(p + 16);
  f32x4 a3 = *(const f32x4*)(p + 20);
  v16bf v;
#pragma unroll
  for (int i = 0; i < 4; ++i) {
    v[i]      = (__bf16)a0[i];
    v[4 + i]  = (__bf16)a1[i];
    v[8 + i]  = (__bf16)a2[i];
    v[12 + i] = (__bf16)a3[i];
  }
  return v;
}

__device__ __forceinline__ v16bf pack_a(const __bf16* p) {
  bf16x8 a0 = *(const bf16x8*)(p);
  bf16x8 a1 = *(const bf16x8*)(p + 16);
  v16bf v;
#pragma unroll
  for (int i = 0; i < 8; ++i) { v[i] = a0[i]; v[8 + i] = a1[i]; }
  return v;
}

__device__ __forceinline__ v16bf pack_b(const float* p) {  // p[0..7], p[16..23]
  v16bf v;
#pragma unroll
  for (int i = 0; i < 16; i += 4) {
    f32x4 t = *(const f32x4*)(p + (i < 8 ? i : i + 8));
    v[i] = (__bf16)t[0]; v[i+1] = (__bf16)t[1];
    v[i+2] = (__bf16)t[2]; v[i+3] = (__bf16)t[3];
  }
  return v;
}

__device__ __forceinline__ v16bf pack_b(const __bf16* p) { // p[0..7], p[16..23]
  bf16x8 a0 = *(const bf16x8*)(p);
  bf16x8 a1 = *(const bf16x8*)(p + 16);
  v16bf v;
#pragma unroll
  for (int i = 0; i < 8; ++i) { v[i] = a0[i]; v[8 + i] = a1[i]; }
  return v;
}

__device__ __forceinline__ v8f wmma_bf16(v16bf a, v16bf b, v8f c) {
  return __builtin_amdgcn_wmma_f32_16x16x32_bf16(false, a, false, b, (short)0, c,
                                                 false, false);
}

// ---------------------------------------------------------------------------
// C[M,N] = A[M,K] * W[N,K]^T + bias[N]   (torch Linear)
// Register-blocked: each wave owns a 16x64 output strip (4 N-tiles), so per
// 32-wide k-step it loads ONE A fragment and four B fragments and issues FOUR
// v_wmma ops -> A traffic amortized 4x, wmma:vmem ratio ~doubled vs 16x16.
// A: fp32 or bf16 row-major.  W: fp32 row-major [N,K] (converted inline).
// Output: fp32 row-major (outf) OR bf16 (outb) row-major / per-batch
// transposed [b][N][seq] when transSeq > 0 (lays V out as V^T so the P*V
// WMMA B-fragments load contiguously along the sequence dim).
// ---------------------------------------------------------------------------
template <typename AT>
__global__ __launch_bounds__(256) void gemm_bias_wmma(
    const AT* __restrict__ A, const float* __restrict__ W,
    const float* __restrict__ bias, __bf16* __restrict__ outb,
    float* __restrict__ outf, int M, int N, int K, int transSeq) {
  const int lane  = threadIdx.x & 31;
  const int wave  = threadIdx.x >> 5;
  const int strip = blockIdx.x * (blockDim.x >> 5) + wave;
  const int tilesM = M >> 4;
  const int tm  = strip % tilesM;
  const int tn4 = strip / tilesM;          // 64-wide N strip index
  if (tn4 >= (N >> 6)) return;
  const int lh = lane >> 4;                // lane half
  const int ln = lane & 15;

  const AT*    arow = A + (size_t)(tm * 16 + ln) * K;
  const float* wrow = W + (size_t)(tn4 * 64 + ln) * K;  // +nt*16*K per N-tile

  v8f acc[4] = {{}, {}, {}, {}};
  for (int k0 = 0; k0 < K; k0 += 32) {
    const v16bf av = pack_a(arow + k0 + (lh << 3));
#pragma unroll
    for (int nt = 0; nt < 4; ++nt) {
      v16bf bv = pack_b(wrow + (size_t)nt * 16 * K + k0 + (lh << 3));
      acc[nt] = wmma_bf16(av, bv, acc[nt]);
    }
  }

#pragma unroll
  for (int nt = 0; nt < 4; ++nt) {
    const int   col = tn4 * 64 + nt * 16 + ln;
    const float bn  = bias[col];
#pragma unroll
    for (int r = 0; r < 8; ++r) {
      const int   row = tm * 16 + r + (lh << 3);
      const float v   = acc[nt][r] + bn;
      if (outf) {
        outf[(size_t)row * N + col] = v;
      } else if (transSeq > 0) {
        const int b = row / transSeq;
        const int s = row % transSeq;
        outb[((size_t)b * N + col) * transSeq + s] = (__bf16)v;
      } else {
        outb[(size_t)row * N + col] = (__bf16)v;
      }
    }
  }
}

// ---------------------------------------------------------------------------
// Attention core: one workgroup = (b, h, 16-query-row tile).
//   Phase 1: S = Q K^T / 8 + bias  -> LDS (16 x 2048 fp32 = 128KB, legal in
//            CDNA5's 320KB/WGP LDS). Q fragments stay in registers for all
//            128 k-tiles.
//   Phase 2: row softmax in LDS; normalized probs streamed to attn out with
//            non-temporal stores (537MB > 192MB L2 -> pure streaming).
//   Phase 3: O = P V via WMMA; A from LDS probs, B from transposed V
//            (contiguous in sequence dim); k-split over waves, LDS combine.
// ---------------------------------------------------------------------------
__global__ __launch_bounds__(256) void attn_kernel(
    const __bf16* __restrict__ Qbf,   // [B*NQ, D] row-major
    const __bf16* __restrict__ Kbf,   // [B*NK, D] row-major
    const __bf16* __restrict__ Vt,    // [B, D, NK]  (transposed V)
    const float* __restrict__ vbias,  // [B, NQ, NK]
    float* __restrict__ attn,         // [B, H, NQ, NK]
    __bf16* __restrict__ Obf) {       // [B*NQ, D]
  extern __shared__ float smem_f[];
  float* S       = smem_f;            // 16*NKS
  float* red     = S + 16 * NKS;      // 256
  float* rowstat = red + 256;         // 32
  float* part    = rowstat + 32;      // 1024

  const int qt = blockIdx.x;          // q tile (16 rows)
  const int h  = blockIdx.y;
  const int b  = blockIdx.z;
  const int lane = threadIdx.x & 31;
  const int wave = threadIdx.x >> 5;
  const int lh = lane >> 4, ln = lane & 15;
  const int qbase = qt * 16;

  // --- Phase 1: scores ---
  const __bf16* qrow =
      Qbf + ((size_t)(b * NQS + qbase + ln)) * DMODEL + h * DHEAD;
  const v16bf aq0 = pack_a(qrow + (lh << 3));        // d = 0..31
  const v16bf aq1 = pack_a(qrow + 32 + (lh << 3));   // d = 32..63

  for (int kt = wave; kt < NKS / 16; kt += 8) {
    const __bf16* krow =
        Kbf + ((size_t)(b * NKS + kt * 16 + ln)) * DMODEL + h * DHEAD;
    v16bf bk0 = pack_b(krow + (lh << 3));
    v16bf bk1 = pack_b(krow + 32 + (lh << 3));
    v8f acc = {};
    acc = wmma_bf16(aq0, bk0, acc);
    acc = wmma_bf16(aq1, bk1, acc);
    const int col = kt * 16 + ln;
#pragma unroll
    for (int r = 0; r < 8; ++r) {
      const int m = r + (lh << 3);
      const float s =
          acc[r] * 0.125f +
          vbias[((size_t)(b * NQS + qbase + m)) * NKS + col];
      S[m * NKS + col] = s;
    }
  }
  __syncthreads();

  // --- Phase 2: softmax over each of the 16 rows (16 threads/row) ---
  const int r = threadIdx.x >> 4;  // row 0..15
  const int j = threadIdx.x & 15;
  float mx = -3.0e38f;
  for (int i = j; i < NKS; i += 16) mx = fmaxf(mx, S[r * NKS + i]);
  red[r * 16 + j] = mx;
  __syncthreads();
  if (j == 0) {
    float m = red[r * 16];
#pragma unroll
    for (int t = 1; t < 16; ++t) m = fmaxf(m, red[r * 16 + t]);
    rowstat[r] = m;
  }
  __syncthreads();
  const float rmax = rowstat[r];
  float sum = 0.f;
  for (int i = j; i < NKS; i += 16) {
    const float e = __expf(S[r * NKS + i] - rmax);
    S[r * NKS + i] = e;
    sum += e;
  }
  red[r * 16 + j] = sum;
  __syncthreads();
  if (j == 0) {
    float s = 0.f;
#pragma unroll
    for (int t = 0; t < 16; ++t) s += red[r * 16 + t];
    rowstat[16 + r] = 1.0f / s;
  }
  __syncthreads();
  const float inv = rowstat[16 + r];
  float* ao = attn + (((size_t)(b * NHEADS + h)) * NQS + qbase) * NKS;
  for (int i = j; i < NKS; i += 16) {
    const float p = S[r * NKS + i] * inv;
    S[r * NKS + i] = p;
    __builtin_nontemporal_store(p, &ao[(size_t)r * NKS + i]);  // stream 537MB
  }
  __syncthreads();

  // --- Phase 3: O = P * V.  wave -> d-tile (wave&3), k-half (wave>>2) ---
  const int dt   = wave & 3;
  const int half = wave >> 2;
  v8f oacc = {};
  const __bf16* vcol =
      Vt + ((size_t)b * DMODEL + h * DHEAD + dt * 16 + ln) * NKS;
  for (int kk0 = half * (NKS / 2); kk0 < (half + 1) * (NKS / 2); kk0 += 32) {
    v16bf ap = pack_a(S + (size_t)ln * NKS + kk0 + (lh << 3));  // LDS fp32->bf16
    v16bf bv = pack_b(vcol + kk0 + (lh << 3));
    oacc = wmma_bf16(ap, bv, oacc);
  }
  if (wave >= 4) {
#pragma unroll
    for (int rr = 0; rr < 8; ++rr)
      part[(((wave - 4) * 32 + lane) << 3) + rr] = oacc[rr];
  }
  __syncthreads();
  if (wave < 4) {
#pragma unroll
    for (int rr = 0; rr < 8; ++rr) {
      const float v = oacc[rr] + part[((wave * 32 + lane) << 3) + rr];
      const int m = rr + (lh << 3);
      Obf[((size_t)(b * NQS + qbase + m)) * DMODEL + h * DHEAD + dt * 16 + ln] =
          (__bf16)v;
    }
  }
}

// ---------------------------------------------------------------------------
extern "C" void kernel_launch(void* const* d_in, const int* in_sizes, int n_in,
                              void* d_out, int out_size, void* d_ws,
                              size_t ws_size, hipStream_t stream) {
  (void)in_sizes; (void)n_in; (void)out_size; (void)ws_size;
  const float* query = (const float*)d_in[0];
  const float* key   = (const float*)d_in[1];
  const float* value = (const float*)d_in[2];
  const float* vbias = (const float*)d_in[3];
  const float* Wq = (const float*)d_in[4];
  const float* bq = (const float*)d_in[5];
  const float* Wk = (const float*)d_in[6];
  const float* bk = (const float*)d_in[7];
  const float* Wv = (const float*)d_in[8];
  const float* bv = (const float*)d_in[9];
  const float* Wo = (const float*)d_in[10];
  const float* bo = (const float*)d_in[11];

  const size_t mat = (size_t)BB * NQS * DMODEL;  // elements per bf16 matrix
  __bf16* Qbf = (__bf16*)d_ws;
  __bf16* Kbf = Qbf + mat;
  __bf16* Vt  = Kbf + mat;
  __bf16* Obf = Vt + mat;

  float* out  = (float*)d_out;                       // [B, NQ, D]
  float* attn = out + (size_t)BB * NQS * DMODEL;     // [B, H, NQ, NK]

  const int M = BB * NQS;                            // 4096
  const int strips = (M / 16) * (DMODEL / 64);       // 4096 strips -> 512 blocks

  gemm_bias_wmma<float><<<strips / 8, 256, 0, stream>>>(
      query, Wq, bq, Qbf, nullptr, M, DMODEL, DMODEL, 0);
  gemm_bias_wmma<float><<<strips / 8, 256, 0, stream>>>(
      key, Wk, bk, Kbf, nullptr, M, DMODEL, DMODEL, 0);
  gemm_bias_wmma<float><<<strips / 8, 256, 0, stream>>>(
      value, Wv, bv, Vt, nullptr, M, DMODEL, DMODEL, NKS);  // V^T layout

  const size_t lds_bytes = (size_t)(16 * NKS + 256 + 32 + 1024) * sizeof(float);
  hipFuncSetAttribute((const void*)attn_kernel,
                      hipFuncAttributeMaxDynamicSharedMemorySize,
                      (int)lds_bytes);  // idempotent, not a stream op
  dim3 agrid(NQS / 16, NHEADS, BB);
  attn_kernel<<<agrid, 256, lds_bytes, stream>>>(Qbf, Kbf, Vt, vbias, attn, Obf);

  gemm_bias_wmma<__bf16><<<strips / 8, 256, 0, stream>>>(
      Obf, Wo, bo, nullptr, out, M, DMODEL, DMODEL, 0);
}
